// EquivariantScalarMLP_69234872812251
// MI455X (gfx1250) — matrix operand order repaired and checked by reading, not compiled, not run on hardware
//
#include <hip/hip_runtime.h>
#include <hip/hip_bf16.h>

// CDNA5 (gfx1250) wave32 WMMA types
typedef __attribute__((ext_vector_type(16))) __bf16 v16bf;
typedef __attribute__((ext_vector_type(8)))  __bf16 v8bf;
typedef __attribute__((ext_vector_type(8)))  float  v8f;

#define WMMA_BF16(a, b, c) \
    __builtin_amdgcn_wmma_f32_16x16x32_bf16(false, (a), false, (b), (short)0, (c), false, false)

// Padded LDS row strides (elements) — avoid 64-bank conflicts while keeping
// every b128/v8bf access 16B-aligned.
#define CONDB_LD 72    // bf16, 144 B/row
#define GBUF_LD  516   // f32
#define HBUF_LD  260   // f32
#define SA_LD    264   // bf16, 528 B/row
#define WBUF_LD  2052  // f32
#define XBUF_LD  260   // f32

// ---------------------------------------------------------------------------
// CDNA5 async global<->LDS (ASYNCcnt) — inline asm, portable across toolchains.
// Low 32 bits of a generic pointer into __shared__ are the wave-relative LDS
// byte offset (ISA 10.2 aperture rule).
// ---------------------------------------------------------------------------
__device__ inline unsigned lds_off(const void* p) {
    return (unsigned)(unsigned long long)p;
}
__device__ inline void async_load_b128(unsigned ldsoff, const void* gaddr) {
    asm volatile("global_load_async_to_lds_b128 %0, %1, off"
                 :: "v"(ldsoff), "v"(gaddr) : "memory");
}
__device__ inline void async_store_b128(void* gaddr, unsigned ldsoff) {
    asm volatile("global_store_async_from_lds_b128 %0, %1, off"
                 :: "v"(gaddr), "v"(ldsoff) : "memory");
}
__device__ inline void wait_async0() {
    asm volatile("s_wait_asynccnt 0x0" ::: "memory");
}

// ---------------------------------------------------------------------------
// A-operand loader: 16x32 bf16 tile from a row-major [M][ldk] bf16 array.
// ISA 7.12.2 (16-bit A 16x32): lane<16 holds row=lane, K {kb..kb+7, kb+16..kb+23};
// lane>=16 holds row=lane-16, K {kb+8..kb+15, kb+24..kb+31}.
// ---------------------------------------------------------------------------
__device__ inline v16bf load_a_tile(const __bf16* base, int ldk, int mtile,
                                    int kbase, int lane) {
    int m    = mtile * 16 + (lane & 15);
    int koff = (lane >> 4) * 8;
    const __bf16* p = base + (size_t)m * ldk + kbase + koff;
    v8bf lo = *(const v8bf*)(p);        // K = kbase+koff .. +7
    v8bf hi = *(const v8bf*)(p + 16);   // K = kbase+16+koff .. +7
    return __builtin_shufflevector(lo, hi, 0, 1, 2, 3, 4, 5, 6, 7,
                                           8, 9, 10, 11, 12, 13, 14, 15);
}

// ---------------------------------------------------------------------------
// B-operand loader: 32x16 bf16 tile from a TRANSPOSED weight Wt[N][K] (K inner).
// Column N = lane&15; lanes 0-15 hold K {kbase..+7, kbase+16..+23}, lanes 16-31
// K {kbase+8..+15, kbase+24..+31} -> two aligned 16B loads (matches A).
// ---------------------------------------------------------------------------
__device__ inline v16bf load_b_tile_t(const __bf16* wt, int ldk, int n0,
                                      int kbase, int lane) {
    int n     = n0 + (lane & 15);
    int koff2 = (lane >> 4) * 8;
    const __bf16* p = wt + (size_t)n * ldk + kbase + koff2;
    v8bf lo = *(const v8bf*)(p);
    v8bf hi = *(const v8bf*)(p + 16);
    return __builtin_shufflevector(lo, hi, 0, 1, 2, 3, 4, 5, 6, 7,
                                           8, 9, 10, 11, 12, 13, 14, 15);
}

// ---------------------------------------------------------------------------
// Weight convert + transpose: out[n*K + k] = (bf16) in[k*N + n]
// ---------------------------------------------------------------------------
__global__ void convT(const float* __restrict__ in, __bf16* __restrict__ out,
                      int K, int N) {
    int total = K * N;
    for (int e = blockIdx.x * blockDim.x + threadIdx.x; e < total;
         e += gridDim.x * blockDim.x) {
        int n = e / K;
        int k = e - n * K;
        out[e] = (__bf16)in[(size_t)k * N + n];
    }
}

// ---------------------------------------------------------------------------
// Kernel 1: FiLM1 -> fc1 -> FiLM2 -> inv (out_scalars) ; store bf16 scalars2.
// 32 rows per block, 256 threads (8 waves). All GEMMs via v_wmma bf16.
// LDS: condb 4.5K | gbuf 64.5K | hbuf 32.5K | sA 16.5K  ~ 118K (padded rows)
// ---------------------------------------------------------------------------
__global__ void k1(const float* __restrict__ features,
                   const float* __restrict__ cond,
                   const __bf16* __restrict__ film1t, const float* __restrict__ film1b,
                   const __bf16* __restrict__ fc1t,
                   const __bf16* __restrict__ film2t, const float* __restrict__ film2b,
                   const __bf16* __restrict__ invt,
                   float* __restrict__ out, __bf16* __restrict__ scal2) {
    extern __shared__ __align__(32) char smem[];
    __bf16* condb = (__bf16*)smem;                     // [32][CONDB_LD]
    float*  gbuf  = (float*)(smem + 4608);             // [32][GBUF_LD]
    float*  hbuf  = (float*)(smem + 4608 + 66048);     // [32][HBUF_LD]
    __bf16* sA    = (__bf16*)(smem + 4608 + 66048 + 33280); // [32][SA_LD]

    const int tid  = threadIdx.x;
    const int lane = tid & 31;
    const int wave = tid >> 5;
    const int rowbase = blockIdx.x * 32;

    // conditioning tile -> bf16 LDS (f32->bf16 convert, VALU path required)
    for (int e = tid; e < 32 * 64; e += 256) {
        int m = e >> 6, k = e & 63;
        condb[m * CONDB_LD + k] = (__bf16)cond[(size_t)(rowbase + m) * 64 + k];
    }
    __syncthreads();

    // gb1 = cond @ film1_w + film1_b   (M=32, N=512, K=64)
    for (int t = wave; t < 64; t += 8) {
        int mtile = t & 1, ntile = t >> 1;
        v8f c = {};
        for (int ks = 0; ks < 2; ++ks) {
            v16bf a = load_a_tile(condb, CONDB_LD, mtile, ks * 32, lane);
            v16bf b = load_b_tile_t(film1t, 64, ntile * 16, ks * 32, lane);
            c = WMMA_BF16(a, b, c);
        }
        int n = ntile * 16 + (lane & 15);
        float bias = film1b[n];
        int mrow = mtile * 16 + (lane >> 4) * 8;
#pragma unroll
        for (int v = 0; v < 8; ++v) gbuf[(mrow + v) * GBUF_LD + n] = c[v] + bias;
    }
    __syncthreads();

    // FiLM1 applied to scalar features -> bf16 A operand
    for (int e = tid; e < 32 * 256; e += 256) {
        int m = e >> 8, k = e & 255;
        float s = features[(size_t)(rowbase + m) * 512 + k];
        sA[m * SA_LD + k] =
            (__bf16)(s * gbuf[m * GBUF_LD + k] + gbuf[m * GBUF_LD + 256 + k]);
    }
    __syncthreads();

    // hbuf = sA @ fc1_w   (M=32, N=256, K=256)
    for (int t = wave; t < 32; t += 8) {
        int mtile = t & 1, ntile = t >> 1;
        v8f c = {};
        for (int ks = 0; ks < 8; ++ks) {
            v16bf a = load_a_tile(sA, SA_LD, mtile, ks * 32, lane);
            v16bf b = load_b_tile_t(fc1t, 256, ntile * 16, ks * 32, lane);
            c = WMMA_BF16(a, b, c);
        }
        int n = ntile * 16 + (lane & 15);
        int mrow = mtile * 16 + (lane >> 4) * 8;
#pragma unroll
        for (int v = 0; v < 8; ++v) hbuf[(mrow + v) * HBUF_LD + n] = c[v];
    }

    // gb2 = cond @ film2_w + film2_b (gbuf readers finished at FiLM1 barrier)
    for (int t = wave; t < 64; t += 8) {
        int mtile = t & 1, ntile = t >> 1;
        v8f c = {};
        for (int ks = 0; ks < 2; ++ks) {
            v16bf a = load_a_tile(condb, CONDB_LD, mtile, ks * 32, lane);
            v16bf b = load_b_tile_t(film2t, 64, ntile * 16, ks * 32, lane);
            c = WMMA_BF16(a, b, c);
        }
        int n = ntile * 16 + (lane & 15);
        float bias = film2b[n];
        int mrow = mtile * 16 + (lane >> 4) * 8;
#pragma unroll
        for (int v = 0; v < 8; ++v) gbuf[(mrow + v) * GBUF_LD + n] = c[v] + bias;
    }
    __syncthreads();

    // FiLM2 -> final scalars (bf16)
    for (int e = tid; e < 32 * 256; e += 256) {
        int m = e >> 8, k = e & 255;
        sA[m * SA_LD + k] = (__bf16)(hbuf[m * HBUF_LD + k] * gbuf[m * GBUF_LD + k] +
                                     gbuf[m * GBUF_LD + 256 + k]);
    }
    __syncthreads();

    // spill bf16 scalars2 for kernel 2 via async LDS->global (ASYNCcnt path),
    // 32 rows x 32 b128 chunks = 1024 chunks, 4 per thread
    for (int c = tid; c < 1024; c += 256) {
        int m = c >> 5, j = c & 31;
        async_store_b128(scal2 + (size_t)rowbase * 256 + m * 256 + j * 8,
                         lds_off(sA + m * SA_LD + j * 8));
    }

    // out_scalars = sA @ inv_w  (M=32, N=128, K=256) -> d_out[:,0:128]
    for (int t = wave; t < 16; t += 8) {
        int mtile = t & 1, ntile = t >> 1;
        v8f c = {};
        for (int ks = 0; ks < 8; ++ks) {
            v16bf a = load_a_tile(sA, SA_LD, mtile, ks * 32, lane);
            v16bf b = load_b_tile_t(invt, 256, ntile * 16, ks * 32, lane);
            c = WMMA_BF16(a, b, c);
        }
        int n = ntile * 16 + (lane & 15);
        int mrow = mtile * 16 + (lane >> 4) * 8;
#pragma unroll
        for (int v = 0; v < 8; ++v)
            out[(size_t)(rowbase + mrow + v) * 384 + n] = c[v];
    }
    wait_async0();   // drain async stores before wave end
}

// ---------------------------------------------------------------------------
// Kernel 2: w = scalars2 @ wemb_w fused with the per-row equivariant einsum.
// 16 rows/block, 8 waves: wave w owns N columns [w*256, w*256+256).
// w tile (16x2052 f32) lives only in LDS -> avoids ~1 GB of HBM traffic.
// LDS: wbuf 128.3K | sA 8.3K | xbuf 16.3K ~ 153K
// ---------------------------------------------------------------------------
__global__ void k2(const float* __restrict__ features,
                   const __bf16* __restrict__ wembt,
                   const __bf16* __restrict__ scal2,
                   float* __restrict__ out) {
    extern __shared__ __align__(32) char smem[];
    float*  wbuf = (float*)smem;                        // [16][WBUF_LD]
    __bf16* sA   = (__bf16*)(smem + 131328);            // [16][SA_LD]
    float*  xbuf = (float*)(smem + 131328 + 8448);      // [16][XBUF_LD]

    const int tid  = threadIdx.x;
    const int lane = tid & 31;
    const int wave = tid >> 5;
    const int rowbase = blockIdx.x * 16;

    // async global->LDS fills (dtype-preserving, ASYNCcnt path)
    // sA: 16 rows x 32 b128 chunks (bf16), 2 per thread
    for (int c = tid; c < 512; c += 256) {
        int m = c >> 5, j = c & 31;
        async_load_b128(lds_off(sA + m * SA_LD + j * 8),
                        scal2 + (size_t)rowbase * 256 + m * 256 + j * 8);
    }
    // xbuf: 16 rows x 64 b128 chunks (f32 equiv features), 4 per thread
    for (int c = tid; c < 1024; c += 256) {
        int m = c >> 6, j = c & 63;
        async_load_b128(lds_off(xbuf + m * XBUF_LD + j * 4),
                        features + (size_t)(rowbase + m) * 512 + 256 + j * 4);
    }
    wait_async0();
    __syncthreads();

    // hoist A tiles (single M tile, 8 K-steps) and reuse across 16 N tiles
    v16bf aT[8];
#pragma unroll
    for (int ks = 0; ks < 8; ++ks) aT[ks] = load_a_tile(sA, SA_LD, 0, ks * 32, lane);

    for (int nt = 0; nt < 16; ++nt) {
        int n0 = (wave * 16 + nt) * 16;
        v8f c = {};
#pragma unroll
        for (int ks = 0; ks < 8; ++ks) {
            v16bf b = load_b_tile_t(wembt, 256, n0, ks * 32, lane);
            c = WMMA_BF16(aT[ks], b, c);
        }
        int n = n0 + (lane & 15);
        int mrow = (lane >> 4) * 8;
#pragma unroll
        for (int v = 0; v < 8; ++v) wbuf[(mrow + v) * WBUF_LD + n] = c[v];
    }
    __syncthreads();

    // per-row einsum: y1 = w1^T @ x1, y2 = w2^T @ x2, scaled by 1/sqrt(32)
    const float norm = 0.17677669529663687f;
    int r = tid >> 4;                       // row 0..15
    const float* wrow = wbuf + r * WBUF_LD;
    const float* xrow = xbuf + r * XBUF_LD;
    float* orow = out + (size_t)(rowbase + r) * 384;
    for (int od = (tid & 15); od < 256; od += 16) {
        float acc = 0.f;
        if (od < 96) {                      // y1: 32x3
            int o = od / 3, d = od - o * 3;
#pragma unroll
            for (int i = 0; i < 32; ++i)
                acc += wrow[i * 32 + o] * xrow[i * 3 + d];
            orow[128 + od] = acc * norm;
        } else {                            // y2: 32x5
            int od2 = od - 96;
            int o = od2 / 5, d = od2 - o * 5;
#pragma unroll
            for (int i = 0; i < 32; ++i)
                acc += wrow[1024 + i * 32 + o] * xrow[96 + i * 5 + d];
            orow[224 + od2] = acc * norm;
        }
    }
}

// ---------------------------------------------------------------------------
extern "C" void kernel_launch(void* const* d_in, const int* in_sizes, int n_in,
                              void* d_out, int out_size, void* d_ws, size_t ws_size,
                              hipStream_t stream) {
    const float* features = (const float*)d_in[0];
    const float* cond     = (const float*)d_in[1];
    const float* film1w   = (const float*)d_in[2];
    const float* film1b   = (const float*)d_in[3];
    const float* fc1w     = (const float*)d_in[4];
    const float* film2w   = (const float*)d_in[5];
    const float* film2b   = (const float*)d_in[6];
    const float* invw     = (const float*)d_in[7];
    const float* wembw    = (const float*)d_in[8];

    char* ws = (char*)d_ws;
    __bf16* film1t = (__bf16*)(ws);                                     // 512x64
    __bf16* fc1t   = (__bf16*)(ws + 65536);                             // 256x256
    __bf16* film2t = (__bf16*)(ws + 65536 + 131072);                    // 512x64
    __bf16* invt   = (__bf16*)(ws + 65536 + 131072 + 65536);            // 128x256
    __bf16* wembt  = (__bf16*)(ws + 65536 + 131072 + 65536 + 65536);    // 2048x256
    __bf16* scal2  = (__bf16*)(ws + 65536 + 131072 + 65536 + 65536 + 1048576);

    size_t lds1 = 4608 + 66048 + 33280 + 16896;   // ~118 KB
    size_t lds2 = 131328 + 8448 + 16640;          // ~153 KB (WGP has 320 KB)
    hipFuncSetAttribute((const void*)k1, hipFuncAttributeMaxDynamicSharedMemorySize, (int)lds1);
    hipFuncSetAttribute((const void*)k2, hipFuncAttributeMaxDynamicSharedMemorySize, (int)lds2);

    // one-time weight convert+transpose to bf16 [N][K]
    convT<<<64, 256, 0, stream>>>(film1w, film1t, 64, 512);
    convT<<<64, 256, 0, stream>>>(fc1w,   fc1t,   256, 256);
    convT<<<64, 256, 0, stream>>>(film2w, film2t, 64, 512);
    convT<<<64, 256, 0, stream>>>(invw,   invt,   256, 128);
    convT<<<512, 256, 0, stream>>>(wembw, wembt,  256, 2048);

    k1<<<65536 / 32, 256, lds1, stream>>>(features, cond, film1t, film1b, fc1t,
                                          film2t, film2b, invt,
                                          (float*)d_out, scal2);
    k2<<<65536 / 16, 256, lds2, stream>>>(features, wembt, scal2, (float*)d_out);
}
